// HeteroGAT_66846870995281
// MI455X (gfx1250) — matrix operand order repaired and checked by reading, not compiled, not run on hardware
//
#include <hip/hip_runtime.h>
#include <hip/hip_bf16.h>

// ---------------------------------------------------------------------------
// Problem constants (match reference)
// ---------------------------------------------------------------------------
#define NP 50000
#define NV 100000
#define ND 20000
#define NE 300000
#define FDIM 128
#define OUTP 8

typedef __attribute__((ext_vector_type(16))) __bf16 v16bf;
typedef __attribute__((ext_vector_type(8)))  float  v8f;

union ABFrag { v16bf v; uint4 q[2]; };

// ---------------------------------------------------------------------------
// Strip-mined WMMA GEMM: H[N x 128] = X[N x 128] @ W[128 x 128]
// X bf16 row-major; Wt bf16 TRANSPOSED (Wt[n*128+k] = W[k][n]).
// Block = 256 threads = 8 waves. The 32KB weight matrix is staged in LDS once
// per block and shared by all 8 waves (B fragments come from ds_load_b128,
// not per-wave L2 hits). Each wave computes one FULL 16x128 row strip
// (8 column tiles, 8 f32 accumulators, 32 WMMAs); A fragments are loaded once
// per k-step and reused across all 8 column tiles.
// ---------------------------------------------------------------------------
__global__ __launch_bounds__(256) void gemm_wmma_bf16out(
    const __bf16* __restrict__ X, const __bf16* __restrict__ Wt,
    __bf16* __restrict__ H, int N)
{
    __shared__ __align__(16) __bf16 Wl[FDIM * FDIM];   // 32 KB

    // cooperative stage: 2048 uint4 / 256 threads = 8 per thread
    {
        const uint4* wg = (const uint4*)Wt;
        uint4* wl = (uint4*)Wl;
#pragma unroll
        for (int i = 0; i < 8; ++i)
            wl[threadIdx.x + i * 256] = wg[threadIdx.x + i * 256];
    }
    __syncthreads();

    const int wave = threadIdx.x >> 5;
    const int lane = threadIdx.x & 31;
    const int tiles = N >> 4;
    const int tm0  = blockIdx.x * 8 + wave;          // row tile
    const int tm   = tm0 < tiles ? tm0 : tiles - 1;  // clamp: keep EXEC all-1s
    const int col  = lane & 15;                      // A row / B col / D col
    const int half = lane >> 4;

    const long rowBase = (long)(tm * 16 + col) * FDIM;
    const __bf16* wbase = Wl + (long)col * FDIM;

    v8f acc[8];
#pragma unroll
    for (int tn = 0; tn < 8; ++tn) acc[tn] = (v8f){};

#pragma unroll
    for (int ks = 0; ks < 4; ++ks) {
        const int kk = ks * 32;
        ABFrag a;
        // A 16x32 bf16: elems 0..15 -> K = kk + 16*half + 0..15 (contig)
        a.q[0] = *(const uint4*)(X + rowBase + kk + half * 16);
        a.q[1] = *(const uint4*)(X + rowBase + kk + half * 16 + 8);
        if (ks < 3)
            __builtin_prefetch(X + rowBase + kk + 32, 0, 3);
#pragma unroll
        for (int tn = 0; tn < 8; ++tn) {
            const __bf16* wcol = wbase + (long)tn * 16 * FDIM;
            ABFrag b;
            // B 32x16 bf16: lane = col, K = kk + 16*half + 0..15 (contig, LDS)
            b.q[0] = *(const uint4*)(wcol + kk + half * 16);
            b.q[1] = *(const uint4*)(wcol + kk + half * 16 + 8);
            acc[tn] = __builtin_amdgcn_wmma_f32_16x16x32_bf16(
                false, a.v, false, b.v, (short)0, acc[tn], false, false);
        }
    }

    if (tm0 < tiles) {
        const long outRow = (long)(tm * 16) * FDIM + col;
#pragma unroll
        for (int tn = 0; tn < 8; ++tn) {
#pragma unroll
            for (int v = 0; v < 8; ++v) {
                const int m = v + 8 * half;          // D: row = vgpr + 8*half
                H[outRow + (long)m * FDIM + tn * 16] = (__bf16)acc[tn][v];
            }
        }
    }
}

// Same strip-mined LDS-staged GEMM, f32 output + bias (hv / hd heads -> d_out).
__global__ __launch_bounds__(256) void gemm_wmma_f32out_bias(
    const __bf16* __restrict__ X, const __bf16* __restrict__ Wt,
    float* __restrict__ H, const float* __restrict__ bias, int N)
{
    __shared__ __align__(16) __bf16 Wl[FDIM * FDIM];

    {
        const uint4* wg = (const uint4*)Wt;
        uint4* wl = (uint4*)Wl;
#pragma unroll
        for (int i = 0; i < 8; ++i)
            wl[threadIdx.x + i * 256] = wg[threadIdx.x + i * 256];
    }
    __syncthreads();

    const int wave = threadIdx.x >> 5;
    const int lane = threadIdx.x & 31;
    const int tiles = N >> 4;
    const int tm0  = blockIdx.x * 8 + wave;
    const int tm   = tm0 < tiles ? tm0 : tiles - 1;
    const int col  = lane & 15;
    const int half = lane >> 4;

    const long rowBase = (long)(tm * 16 + col) * FDIM;
    const __bf16* wbase = Wl + (long)col * FDIM;

    v8f acc[8];
#pragma unroll
    for (int tn = 0; tn < 8; ++tn) acc[tn] = (v8f){};

#pragma unroll
    for (int ks = 0; ks < 4; ++ks) {
        const int kk = ks * 32;
        ABFrag a;
        a.q[0] = *(const uint4*)(X + rowBase + kk + half * 16);
        a.q[1] = *(const uint4*)(X + rowBase + kk + half * 16 + 8);
#pragma unroll
        for (int tn = 0; tn < 8; ++tn) {
            const __bf16* wcol = wbase + (long)tn * 16 * FDIM;
            ABFrag b;
            b.q[0] = *(const uint4*)(wcol + kk + half * 16);
            b.q[1] = *(const uint4*)(wcol + kk + half * 16 + 8);
            acc[tn] = __builtin_amdgcn_wmma_f32_16x16x32_bf16(
                false, a.v, false, b.v, (short)0, acc[tn], false, false);
        }
    }

    if (tm0 < tiles) {
        const long outRow = (long)(tm * 16) * FDIM + col;
#pragma unroll
        for (int tn = 0; tn < 8; ++tn) {
            const float bv = bias[tn * 16 + col];
#pragma unroll
            for (int v = 0; v < 8; ++v) {
                const int m = v + 8 * half;
                H[outRow + (long)m * FDIM + tn * 16] = acc[tn][v] + bv;
            }
        }
    }
}

// ---------------------------------------------------------------------------
// Small helper kernels
// ---------------------------------------------------------------------------
__global__ void f32_to_bf16_k(const float* __restrict__ in, __bf16* __restrict__ out, long n) {
    long i = (long)blockIdx.x * blockDim.x + threadIdx.x;
    if (i < n) out[i] = (__bf16)in[i];
}

__global__ void relu_f32_to_bf16_k(const float* __restrict__ in, __bf16* __restrict__ out, long n) {
    long i = (long)blockIdx.x * blockDim.x + threadIdx.x;
    if (i < n) { float v = in[i]; out[i] = (__bf16)(v > 0.f ? v : 0.f); }
}

// acc[i] = b0[i&127] (+ b1[i&127])  -- bias folded into dst accumulator
__global__ void init_acc_k(float* __restrict__ acc, const float* __restrict__ b0,
                           const float* __restrict__ b1, long n) {
    long i = (long)blockIdx.x * blockDim.x + threadIdx.x;
    if (i < n) {
        int f = (int)(i & (FDIM - 1));
        float v = b0[f];
        if (b1) v += b1[f];
        acc[i] = v;
    }
}

__global__ void init_mz_k(unsigned* __restrict__ m, float* __restrict__ z, int n) {
    int i = blockIdx.x * blockDim.x + threadIdx.x;
    if (i < n) { m[i] = 0u; z[i] = 0.f; }
}

// c[k] = sum_h W[k*128+h] * a[h]   (folds projection into score vector)
__global__ void combine_wa_k(const float* __restrict__ W, const float* __restrict__ a,
                             float* __restrict__ c) {
    int k = threadIdx.x;
    if (k < FDIM) {
        float s = 0.f;
        const float* row = W + (long)k * FDIM;
#pragma unroll 8
        for (int h = 0; h < FDIM; ++h) s += row[h] * a[h];
        c[k] = s;
    }
}

// Wt[n*128+k] = bf16(W[k*128+n])
__global__ void transpose_to_bf16_k(const float* __restrict__ W, __bf16* __restrict__ Wt) {
    int i = blockIdx.x * blockDim.x + threadIdx.x; // 16384
    int n = i >> 7, k = i & (FDIM - 1);
    Wt[(long)n * FDIM + k] = (__bf16)W[(long)k * FDIM + n];
}

// s[i] = sum_k X[i][k] * c[k]   (bf16 feats, f32 score)
__global__ void node_scores_k(const __bf16* __restrict__ X, const float* __restrict__ c,
                              float* __restrict__ s, int N) {
    int i = blockIdx.x * blockDim.x + threadIdx.x;
    if (i < N) {
        const __bf16* row = X + (long)i * FDIM;
        float acc = 0.f;
#pragma unroll 8
        for (int k = 0; k < FDIM; ++k) acc += (float)row[k] * c[k];
        s[i] = acc;
    }
}

// ---------------------------------------------------------------------------
// Order-preserving float<->uint encoding for atomicMax on signed floats
// ---------------------------------------------------------------------------
__device__ __forceinline__ unsigned enc_f(float f) {
    unsigned u = __float_as_uint(f);
    return (u & 0x80000000u) ? ~u : (u | 0x80000000u);
}
__device__ __forceinline__ float dec_f(unsigned u) {
    return (u & 0x80000000u) ? __uint_as_float(u & 0x7FFFFFFFu) : __uint_as_float(~u);
}

__global__ void edge_logits_k(const int* __restrict__ src, const int* __restrict__ dst,
                              const float* __restrict__ ssrc, const float* __restrict__ sdst,
                              float* __restrict__ ebuf, unsigned* __restrict__ m, int E) {
    int e = blockIdx.x * blockDim.x + threadIdx.x;
    if (e < E) {
        float x = ssrc[src[e]] + sdst[dst[e]];
        x = (x > 0.f) ? x : 0.2f * x;            // LeakyReLU
        ebuf[e] = x;
        atomicMax(&m[dst[e]], enc_f(x));
    }
}

__global__ void edge_weights_k(const int* __restrict__ dst, const float* __restrict__ ebuf,
                               const unsigned* __restrict__ m, float* __restrict__ wbuf,
                               float* __restrict__ z, int E) {
    int e = blockIdx.x * blockDim.x + threadIdx.x;
    if (e < E) {
        int d = dst[e];
        float w = expf(ebuf[e] - dec_f(m[d]));
        wbuf[e] = w;
        atomicAdd(&z[d], w);
    }
}

// 256 threads = 2 edges per block: acc[dst][f] += alpha * hs[src][f]
__global__ __launch_bounds__(256) void edge_scatter_k(
    const int* __restrict__ src, const int* __restrict__ dst,
    const float* __restrict__ wbuf, const float* __restrict__ z,
    const __bf16* __restrict__ hs, float* __restrict__ acc, int E) {
    int e = blockIdx.x * 2 + (threadIdx.x >> 7);
    if (e >= E) return;
    int f = threadIdx.x & (FDIM - 1);
    int d = dst[e];
    float alpha = wbuf[e] / z[d];
    float v = alpha * (float)hs[(long)src[e] * FDIM + f];
    atomicAdd(&acc[(long)d * FDIM + f], v);
}

// patient head: out[i*8+o] = bh[o] + sum_k relu(acc[i][k]) * Wh[k*8+o]
__global__ void head_p_k(const float* __restrict__ accp, const float* __restrict__ Wh,
                         const float* __restrict__ bh, float* __restrict__ out, int N) {
    int i = blockIdx.x * blockDim.x + threadIdx.x;
    if (i < N * OUTP) {
        int n = i >> 3, o = i & 7;
        const float* row = accp + (long)n * FDIM;
        float s = bh[o];
#pragma unroll 8
        for (int k = 0; k < FDIM; ++k) {
            float x = row[k];
            x = x > 0.f ? x : 0.f;
            s += x * Wh[k * OUTP + o];
        }
        out[i] = s;
    }
}

// ---------------------------------------------------------------------------
// Launch
// ---------------------------------------------------------------------------
static inline int cdiv(long a, int b) { return (int)((a + b - 1) / b); }

extern "C" void kernel_launch(void* const* d_in, const int* in_sizes, int n_in,
                              void* d_out, int out_size, void* d_ws, size_t ws_size,
                              hipStream_t stream) {
    (void)in_sizes; (void)n_in; (void)out_size; (void)ws_size;

    const float* xP  = (const float*)d_in[0];
    const float* xV  = (const float*)d_in[1];
    const float* xD  = (const float*)d_in[2];
    const float* Wsrc[2] = { (const float*)d_in[3], (const float*)d_in[8]  };
    const float* Wdst[2] = { (const float*)d_in[4], (const float*)d_in[9]  };
    const float* Asrc[2] = { (const float*)d_in[5], (const float*)d_in[10] };
    const float* Adst[2] = { (const float*)d_in[6], (const float*)d_in[11] };
    const float* Bias[2] = { (const float*)d_in[7], (const float*)d_in[12] };
    const float* WhP = (const float*)d_in[13];
    const float* bhP = (const float*)d_in[14];
    const float* WhV = (const float*)d_in[15];
    const float* bhV = (const float*)d_in[16];
    const float* WhD = (const float*)d_in[17];
    const float* bhD = (const float*)d_in[18];
    const int* eiPV = (const int*)d_in[19];
    const int* eiVP = (const int*)d_in[20];
    const int* eiVD = (const int*)d_in[21];
    const int* eiDV = (const int*)d_in[22];
    float* out = (float*)d_out;

    // ---- workspace carve ----
    char* wsp = (char*)d_ws; size_t off = 0;
    auto walloc = [&](size_t bytes) -> void* {
        void* p = wsp + off; off += (bytes + 255) & ~(size_t)255; return p;
    };
    __bf16* fP  = (__bf16*)walloc((size_t)NP * FDIM * 2);
    __bf16* fV  = (__bf16*)walloc((size_t)NV * FDIM * 2);
    __bf16* fD  = (__bf16*)walloc((size_t)ND * FDIM * 2);
    __bf16* hs  = (__bf16*)walloc((size_t)NV * FDIM * 2);   // max src count
    float*  accP = (float*)walloc((size_t)NP * FDIM * 4);
    float*  accV = (float*)walloc((size_t)NV * FDIM * 4);
    float*  accD = (float*)walloc((size_t)ND * FDIM * 4);
    __bf16* Wt  = (__bf16*)walloc((size_t)FDIM * FDIM * 2);
    float*  cS  = (float*)walloc(FDIM * 4);
    float*  cD  = (float*)walloc(FDIM * 4);
    float*  sS  = (float*)walloc((size_t)NV * 4);
    float*  sDs = (float*)walloc((size_t)NV * 4);
    unsigned* mU = (unsigned*)walloc((size_t)NV * 4);
    float*  zB  = (float*)walloc((size_t)NV * 4);
    float*  eB  = (float*)walloc((size_t)NE * 4);
    float*  wB  = (float*)walloc((size_t)NE * 4);

    struct RelDesc { int r; const __bf16* xs; int Ns; const __bf16* xd; int Nd;
                     const int* ei; float* acc; };

    for (int L = 0; L < 2; ++L) {
        // node features -> bf16 (layer 0: raw inputs; layer 1: relu of layer-0 out)
        if (L == 0) {
            f32_to_bf16_k<<<cdiv((long)NP * FDIM, 256), 256, 0, stream>>>(xP, fP, (long)NP * FDIM);
            f32_to_bf16_k<<<cdiv((long)NV * FDIM, 256), 256, 0, stream>>>(xV, fV, (long)NV * FDIM);
            f32_to_bf16_k<<<cdiv((long)ND * FDIM, 256), 256, 0, stream>>>(xD, fD, (long)ND * FDIM);
        } else {
            relu_f32_to_bf16_k<<<cdiv((long)NP * FDIM, 256), 256, 0, stream>>>(accP, fP, (long)NP * FDIM);
            relu_f32_to_bf16_k<<<cdiv((long)NV * FDIM, 256), 256, 0, stream>>>(accV, fV, (long)NV * FDIM);
            relu_f32_to_bf16_k<<<cdiv((long)ND * FDIM, 256), 256, 0, stream>>>(accD, fD, (long)ND * FDIM);
        }
        // dst accumulators initialized with (summed) relation biases
        init_acc_k<<<cdiv((long)NV * FDIM, 256), 256, 0, stream>>>(accV, Bias[L] + 0 * FDIM, Bias[L] + 3 * FDIM, (long)NV * FDIM);
        init_acc_k<<<cdiv((long)NP * FDIM, 256), 256, 0, stream>>>(accP, Bias[L] + 1 * FDIM, (const float*)nullptr, (long)NP * FDIM);
        init_acc_k<<<cdiv((long)ND * FDIM, 256), 256, 0, stream>>>(accD, Bias[L] + 2 * FDIM, (const float*)nullptr, (long)ND * FDIM);

        const RelDesc rel[4] = {
            { 0, fP, NP, fV, NV, eiPV, accV },   // patient -> visit
            { 1, fV, NV, fP, NP, eiVP, accP },   // visit   -> patient
            { 2, fV, NV, fD, ND, eiVD, accD },   // visit   -> drug
            { 3, fD, ND, fV, NV, eiDV, accV },   // drug    -> visit
        };

        for (int t = 0; t < 4; ++t) {
            const RelDesc& R = rel[t];
            const float* Ws = Wsrc[L] + (long)R.r * FDIM * FDIM;
            const float* Wd = Wdst[L] + (long)R.r * FDIM * FDIM;
            // fold W @ a into per-feature score vectors
            combine_wa_k<<<1, FDIM, 0, stream>>>(Ws, Asrc[L] + R.r * FDIM, cS);
            combine_wa_k<<<1, FDIM, 0, stream>>>(Wd, Adst[L] + R.r * FDIM, cD);
            // transposed bf16 weights for WMMA B operand
            transpose_to_bf16_k<<<FDIM * FDIM / 256, 256, 0, stream>>>(Ws, Wt);
            // per-node attention scores
            node_scores_k<<<cdiv(R.Ns, 256), 256, 0, stream>>>(R.xs, cS, sS, R.Ns);
            node_scores_k<<<cdiv(R.Nd, 256), 256, 0, stream>>>(R.xd, cD, sDs, R.Nd);
            // hs = xs @ W_src   (WMMA, LDS-staged weights, 16x128 strip/wave)
            gemm_wmma_bf16out<<<cdiv(R.Ns / 16, 8), 256, 0, stream>>>(R.xs, Wt, hs, R.Ns);
            // segment-stable softmax over dst
            init_mz_k<<<cdiv(R.Nd, 256), 256, 0, stream>>>(mU, zB, R.Nd);
            edge_logits_k<<<cdiv(NE, 256), 256, 0, stream>>>(R.ei, R.ei + NE, sS, sDs, eB, mU, NE);
            edge_weights_k<<<cdiv(NE, 256), 256, 0, stream>>>(R.ei + NE, eB, mU, wB, zB, NE);
            // weighted scatter into dst accumulator
            edge_scatter_k<<<cdiv(NE, 2), 256, 0, stream>>>(R.ei, R.ei + NE, wB, zB, hs, R.acc, NE);
        }
    }

    // ---- output heads ----
    // hp: [NP x 8] at out + 0  (scalar head; tiny)
    head_p_k<<<cdiv((long)NP * OUTP, 256), 256, 0, stream>>>(accP, WhP, bhP, out, NP);
    // hv: [NV x 128] at out + NP*8  (WMMA head on relu(accV))
    relu_f32_to_bf16_k<<<cdiv((long)NV * FDIM, 256), 256, 0, stream>>>(accV, fV, (long)NV * FDIM);
    transpose_to_bf16_k<<<FDIM * FDIM / 256, 256, 0, stream>>>(WhV, Wt);
    gemm_wmma_f32out_bias<<<cdiv(NV / 16, 8), 256, 0, stream>>>(fV, Wt, out + (long)NP * OUTP, bhV, NV);
    // hd: [ND x 128] after hv
    relu_f32_to_bf16_k<<<cdiv((long)ND * FDIM, 256), 256, 0, stream>>>(accD, fD, (long)ND * FDIM);
    transpose_to_bf16_k<<<FDIM * FDIM / 256, 256, 0, stream>>>(WhD, Wt);
    gemm_wmma_f32out_bias<<<cdiv(ND / 16, 8), 256, 0, stream>>>(
        fD, Wt, out + (long)NP * OUTP + (long)NV * FDIM, bhD, ND);
}
